// DecoderRNN_21406117003359
// MI455X (gfx1250) — matrix operand order repaired and checked by reading, not compiled, not run on hardware
//
#include <hip/hip_runtime.h>
#include <cmath>

// ---- problem dims (fixed by the reference) ----
#define B_ 64
#define T_ 32
#define E_ 512
#define H_ 1024
#define V_ 32000

typedef __attribute__((ext_vector_type(16))) _Float16 v16h;
typedef __attribute__((ext_vector_type(8)))  _Float16 v8h;
typedef __attribute__((ext_vector_type(8)))  float    v8f;
typedef int v4i __attribute__((vector_size(4 * sizeof(int))));

// ---------------------------------------------------------------------------
// CDNA5 async global->LDS copy (ASYNCcnt-tracked), with compile-safe fallback.
// Builtin signature (probed): (global v4i*, shared v4i*, int offset, int cpol)
// ---------------------------------------------------------------------------
#if defined(__has_builtin)
#if __has_builtin(__builtin_amdgcn_global_load_async_to_lds_b128) && \
    __has_builtin(__builtin_amdgcn_s_wait_asynccnt)
#define USE_ASYNC_LDS 1
#endif
#endif

__device__ inline void copy16_g2l(const _Float16* g, _Float16* l) {
#ifdef USE_ASYNC_LDS
    __builtin_amdgcn_global_load_async_to_lds_b128(
        (__attribute__((address_space(1))) v4i*)g,
        (__attribute__((address_space(3))) v4i*)l, 0, 0);
#else
    *(v8h*)l = *(const v8h*)g;
#endif
}

__device__ inline void async_wait0() {
#ifdef USE_ASYNC_LDS
    __builtin_amdgcn_s_wait_asynccnt(0);
#endif
}

// ---------------------------------------------------------------------------
// f32 -> f16 conversion (grid-stride)
// ---------------------------------------------------------------------------
__global__ void cvt_f32_to_f16_kernel(const float* __restrict__ src,
                                      _Float16* __restrict__ dst, long n) {
    long i = (long)blockIdx.x * blockDim.x + threadIdx.x;
    long stride = (long)gridDim.x * blockDim.x;
    for (; i < n; i += stride) dst[i] = (_Float16)src[i];
}

// ---------------------------------------------------------------------------
// Embedding gather + ReLU -> f16, layout x16[t*B + b][E]
// ---------------------------------------------------------------------------
__global__ void embed_relu_kernel(const int* __restrict__ target,
                                  const float* __restrict__ emb,
                                  _Float16* __restrict__ x16) {
    int m = blockIdx.x;            // m = t*B + b
    int t = m / B_;
    int b = m % B_;
    int token = (t == 0) ? 0 : target[b * T_ + (t - 1)];
    const float* e = emb + (size_t)token * E_;
    _Float16* dst = x16 + (size_t)m * E_;
    for (int j = threadIdx.x; j < E_; j += blockDim.x) {
        float v = e[j];
        dst[j] = (_Float16)(v > 0.f ? v : 0.f);
    }
}

__global__ void init_h_kernel(const float* __restrict__ enc_h,
                              float* __restrict__ h32,
                              _Float16* __restrict__ h16) {
    int i = blockIdx.x * blockDim.x + threadIdx.x;
    if (i < B_ * H_) {
        float v = enc_h[i];
        h32[i] = v;
        h16[i] = (_Float16)v;
    }
}

// ---------------------------------------------------------------------------
// WMMA operand assembly (this file's contiguous k order, not the ISA table order; wave32).
//  A lane l -> row M=l%16; elems: K=(l/16)*16+i
//  B lane l -> col N=l%16 of B (row of Bw=[N][K]); elems: K=(l/16)*16+i
// ---------------------------------------------------------------------------
__device__ inline v16h pack_a(const _Float16* p, int half) {
    v8h a0 = *(const v8h*)(p + half * 16);
    v8h a1 = *(const v8h*)(p + half * 16 + 8);
    v16h a;
#pragma unroll
    for (int i = 0; i < 8; ++i) { a[i] = a0[i]; a[i + 8] = a1[i]; }
    return a;
}
__device__ inline v16h pack_b(const _Float16* p, int half) {
    v8h b0 = *(const v8h*)(p + half * 16);
    v8h b1 = *(const v8h*)(p + half * 16 + 8);
    v16h b;
#pragma unroll
    for (int i = 0; i < 8; ++i) { b[i] = b0[i]; b[i + 8] = b1[i]; }
    return b;
}

// ---------------------------------------------------------------------------
// Fused GRU step: one wave per (batch-tile, hidden-tile).
// Computes gh tiles for gates r/z/n (3 accumulators, shared A operand),
// then applies gate nonlinearity and writes h32 / h16(next) / hs16.
// Grid: (H/16, B/16), block = 32 (one wave).
// ---------------------------------------------------------------------------
__global__ __launch_bounds__(32) void gru_step_fused(
    const _Float16* __restrict__ hin,   // [B][H] f16 (state t)
    const _Float16* __restrict__ whh,   // [3H][H] f16
    const float* __restrict__ b_hh,     // [3H]
    const float* __restrict__ gi,       // [T*B][3H] (i gates + b_ih)
    float* __restrict__ h32,            // [B][H] f32 state (in/out)
    _Float16* __restrict__ hout,        // [B][H] f16 (state t+1)
    _Float16* __restrict__ hs16,        // [b*T+t][H]
    int t) {
    const int itile = blockIdx.x;       // hidden tile
    const int mtile = blockIdx.y;       // batch tile
    const int lane = threadIdx.x;
    const int half = lane >> 4;
    const int lm   = lane & 15;

    const _Float16* arow  = hin + (size_t)(mtile * 16 + lm) * H_;
    const _Float16* browR = whh + (size_t)(0 * H_ + itile * 16 + lm) * H_;
    const _Float16* browZ = whh + (size_t)(1 * H_ + itile * 16 + lm) * H_;
    const _Float16* browN = whh + (size_t)(2 * H_ + itile * 16 + lm) * H_;

    v8f accR = {}, accZ = {}, accN = {};
#pragma unroll 2
    for (int k0 = 0; k0 < H_; k0 += 32) {
        v16h a  = pack_a(arow + k0, half);
        v16h bR = *(const v16h*)(browR + k0 + half * 16);
        v16h bZ = *(const v16h*)(browZ + k0 + half * 16);
        v16h bN = *(const v16h*)(browN + k0 + half * 16);
        accR = __builtin_amdgcn_wmma_f32_16x16x32_f16(false, a, false, bR, (short)0, accR, false, false);
        accZ = __builtin_amdgcn_wmma_f32_16x16x32_f16(false, a, false, bZ, (short)0, accZ, false, false);
        accN = __builtin_amdgcn_wmma_f32_16x16x32_f16(false, a, false, bN, (short)0, accN, false, false);
    }

    const int i = itile * 16 + lm;      // hidden index
    const float br = b_hh[i], bz = b_hh[H_ + i], bn = b_hh[2 * H_ + i];
#pragma unroll
    for (int r = 0; r < 8; ++r) {
        int b = mtile * 16 + half * 8 + r;          // batch index
        const float* gir = gi + ((size_t)t * B_ + b) * (3 * H_);
        float ghr = accR[r] + br;
        float ghz = accZ[r] + bz;
        float ghn = accN[r] + bn;
        float rg = 1.f / (1.f + expf(-(gir[i] + ghr)));
        float zg = 1.f / (1.f + expf(-(gir[H_ + i] + ghz)));
        float ng = tanhf(gir[2 * H_ + i] + rg * ghn);
        size_t hidx = (size_t)b * H_ + i;
        float hv = (1.f - zg) * ng + zg * h32[hidx];
        h32[hidx]  = hv;
        hout[hidx] = (_Float16)hv;
        hs16[((size_t)b * T_ + t) * H_ + i] = (_Float16)hv;
    }
}

// ---------------------------------------------------------------------------
// LDS-tiled, async double-buffered WMMA GEMM.
// Block = 256 threads (8 waves), block tile 128x128, K-chunk 64.
// Each wave owns a 16(M) x 128(N) strip: 8 accumulators, A reused across 8
// WMMAs per k-step. Requires M%128==0, N%128==0, K%64==0.
// ---------------------------------------------------------------------------
#define BM 128
#define BN 128
#define BK 64
#define KSTRIDE (BK + 8)   // +8 f16 pad -> 36-dword row stride, conflict-free

__global__ __launch_bounds__(256) void gemm_f16_wmma_tiled(
    const _Float16* __restrict__ A, const _Float16* __restrict__ Bw,
    const float* __restrict__ bias, float* __restrict__ C,
    int M, int N, int K) {
    __shared__ _Float16 sA[2][BM * KSTRIDE];   // 2 * 18432 B
    __shared__ _Float16 sB[2][BN * KSTRIDE];   // 2 * 18432 B

    const int tid  = threadIdx.x;
    const int lane = tid & 31;
    const int wm   = tid >> 5;        // m-wave: 0..7
    const int half = lane >> 4;
    const int lm   = lane & 15;

    const size_t mbase = (size_t)blockIdx.y * BM;
    const size_t nbase = (size_t)blockIdx.x * BN;

    // async loader for one K-chunk into buffer `buf`
    auto load_chunk = [&](int buf, int k0) {
#pragma unroll
        for (int i = 0; i < (BM * BK) / (8 * 256); ++i) {   // 4 ops: A 128x64
            int o = tid + i * 256;
            int row = o >> 3, seg = o & 7;
            copy16_g2l(A + (mbase + row) * K + k0 + seg * 8,
                       &sA[buf][row * KSTRIDE + seg * 8]);
        }
#pragma unroll
        for (int i = 0; i < (BN * BK) / (8 * 256); ++i) {   // 4 ops: B 128x64
            int o = tid + i * 256;
            int row = o >> 3, seg = o & 7;
            copy16_g2l(Bw + (nbase + row) * K + k0 + seg * 8,
                       &sB[buf][row * KSTRIDE + seg * 8]);
        }
    };

    v8f acc[8];
#pragma unroll
    for (int s = 0; s < 8; ++s) acc[s] = (v8f){};

    const int NC = K / BK;
    load_chunk(0, 0);
    async_wait0();
    __syncthreads();

    for (int c = 0; c < NC; ++c) {
        const int cur = c & 1;
        if (c + 1 < NC) load_chunk(1 - cur, (c + 1) * BK);   // overlap with compute

        const _Float16* sa = &sA[cur][(wm * 16 + lm) * KSTRIDE];
        const _Float16* sb = &sB[cur][0];
#pragma unroll
        for (int kk = 0; kk < BK; kk += 32) {
            v16h a = pack_a(sa + kk, half);
#pragma unroll
            for (int s = 0; s < 8; ++s) {
                v16h b = pack_b(sb + (s * 16 + lm) * KSTRIDE + kk, half);
                acc[s] = __builtin_amdgcn_wmma_f32_16x16x32_f16(
                    false, a, false, b, (short)0, acc[s], false, false);
            }
        }
        async_wait0();                // next chunk resident
        __syncthreads();              // all waves done reading `cur`
    }

    const int mrow = (int)mbase + wm * 16 + half * 8;
#pragma unroll
    for (int s = 0; s < 8; ++s) {
        int n = (int)nbase + s * 16 + lm;
        float bv = bias ? bias[n] : 0.f;
#pragma unroll
        for (int r = 0; r < 8; ++r)
            C[(size_t)(mrow + r) * N + n] = acc[s][r] + bv;
    }
}

// ---------------------------------------------------------------------------
// Per-row online log-sum-exp over V; one 256-thread block per row.
// ---------------------------------------------------------------------------
__global__ __launch_bounds__(256) void row_lse_kernel(
    const float* __restrict__ C, float* __restrict__ lse) {
    __shared__ float sm[256];
    __shared__ float ss[256];
    int row = blockIdx.x;
    const float* x = C + (size_t)row * V_;
    float m = -3.0e38f, s = 0.f;
    for (int j = threadIdx.x; j < V_; j += 256) {
        float v = x[j];
        if (v > m) { s = s * expf(m - v) + 1.f; m = v; }
        else       { s += expf(v - m); }
    }
    sm[threadIdx.x] = m;
    ss[threadIdx.x] = s;
    __syncthreads();
    for (int off = 128; off > 0; off >>= 1) {
        if (threadIdx.x < off) {
            float m2 = sm[threadIdx.x + off], s2 = ss[threadIdx.x + off];
            float m1 = sm[threadIdx.x],        s1 = ss[threadIdx.x];
            float M  = m1 > m2 ? m1 : m2;
            sm[threadIdx.x] = M;
            ss[threadIdx.x] = s1 * expf(m1 - M) + s2 * expf(m2 - M);
        }
        __syncthreads();
    }
    if (threadIdx.x == 0) lse[row] = sm[0] + logf(ss[0]);
}

__global__ __launch_bounds__(256) void normalize_kernel(
    float* __restrict__ C, const float* __restrict__ lse) {
    int row = blockIdx.y;
    int j = blockIdx.x * 256 + threadIdx.x;
    if (j < V_) C[(size_t)row * V_ + j] -= lse[row];
}

__global__ void copy_hT_kernel(const float* __restrict__ h32,
                               float* __restrict__ out) {
    int i = blockIdx.x * blockDim.x + threadIdx.x;
    if (i < B_ * H_) out[i] = h32[i];
}

// ---------------------------------------------------------------------------
// Host launcher
// ---------------------------------------------------------------------------
extern "C" void kernel_launch(void* const* d_in, const int* in_sizes, int n_in,
                              void* d_out, int out_size, void* d_ws, size_t ws_size,
                              hipStream_t stream) {
    (void)in_sizes; (void)n_in; (void)out_size; (void)ws_size;
    const float* enc_h  = (const float*)d_in[1];
    const int*   target = (const int*)d_in[2];
    const float* emb    = (const float*)d_in[3];
    const float* w_ih   = (const float*)d_in[4];
    const float* w_hh   = (const float*)d_in[5];
    const float* b_ih   = (const float*)d_in[6];
    const float* b_hh   = (const float*)d_in[7];
    const float* w_out  = (const float*)d_in[8];
    const float* b_out  = (const float*)d_in[9];

    float* out = (float*)d_out;
    float* logits = out;                                   // [B*T][V]
    float* out_hT = out + (size_t)B_ * T_ * V_;            // [B*H]

    char* ws = (char*)d_ws;
    size_t off = 0;
    auto alloc = [&](size_t bytes) {
        char* p = ws + off;
        off += (bytes + 255) & ~(size_t)255;
        return p;
    };
    _Float16* x16    = (_Float16*)alloc((size_t)T_ * B_ * E_ * 2);       // [T*B][E]
    _Float16* wih16  = (_Float16*)alloc((size_t)3 * H_ * E_ * 2);        // [3H][E]
    _Float16* whh16  = (_Float16*)alloc((size_t)3 * H_ * H_ * 2);        // [3H][H]
    _Float16* wout16 = (_Float16*)alloc((size_t)V_ * H_ * 2);            // [V][H]
    float*    gi     = (float*)   alloc((size_t)T_ * B_ * 3 * H_ * 4);   // [T*B][3H]
    float*    h32    = (float*)   alloc((size_t)B_ * H_ * 4);
    _Float16* h16a   = (_Float16*)alloc((size_t)B_ * H_ * 2);
    _Float16* h16b   = (_Float16*)alloc((size_t)B_ * H_ * 2);
    _Float16* hs16   = (_Float16*)alloc((size_t)B_ * T_ * H_ * 2);       // [b*T+t][H]
    float*    lse    = (float*)   alloc((size_t)B_ * T_ * 4);
    _Float16* hbuf[2] = { h16a, h16b };

    // 1) weight conversions to f16
    cvt_f32_to_f16_kernel<<<2048, 256, 0, stream>>>(w_ih, wih16, (long)3 * H_ * E_);
    cvt_f32_to_f16_kernel<<<4096, 256, 0, stream>>>(w_hh, whh16, (long)3 * H_ * H_);
    cvt_f32_to_f16_kernel<<<16384, 256, 0, stream>>>(w_out, wout16, (long)V_ * H_);

    // 2) embedding gather + relu (x laid out [t*B+b][E])
    embed_relu_kernel<<<T_ * B_, 256, 0, stream>>>(target, emb, x16);

    // 3) h0
    init_h_kernel<<<(B_ * H_) / 256, 256, 0, stream>>>(enc_h, h32, hbuf[0]);

    // 4) all input gates at once: gi = x16 @ w_ih^T + b_ih (tiled, async LDS)
    gemm_f16_wmma_tiled<<<dim3(3 * H_ / BN, T_ * B_ / BM), 256, 0, stream>>>(
        x16, wih16, b_ih, gi, T_ * B_, 3 * H_, E_);

    // 5) sequential GRU: one fused WMMA+gate kernel per step
    for (int t = 0; t < T_; ++t) {
        gru_step_fused<<<dim3(H_ / 16, B_ / 16), 32, 0, stream>>>(
            hbuf[t & 1], whh16, b_hh, gi, h32, hbuf[(t + 1) & 1], hs16, t);
    }

    // 6) output projection: logits = hs16 @ w_out^T + b_out (tiled, async LDS)
    gemm_f16_wmma_tiled<<<dim3(V_ / BN, (B_ * T_) / BM), 256, 0, stream>>>(
        hs16, wout16, b_out, logits, B_ * T_, V_, H_);

    // 7) log-softmax
    row_lse_kernel<<<B_ * T_, 256, 0, stream>>>(logits, lse);
    normalize_kernel<<<dim3((V_ + 255) / 256, B_ * T_), 256, 0, stream>>>(logits, lse);

    // 8) hT
    copy_hT_kernel<<<(B_ * H_) / 256, 256, 0, stream>>>(h32, out_hT);
}
